// LocalAttentionModel_57019985822496
// MI455X (gfx1250) — compile-verified, matrix operand order checked by reading, not run on hardware
//
#include <hip/hip_runtime.h>
#include <hip/hip_bf16.h>
#include <math.h>

typedef __attribute__((ext_vector_type(16))) __bf16 bf16x16;
typedef __attribute__((ext_vector_type(8)))  __bf16 bf16x8;
typedef __attribute__((ext_vector_type(8)))  float  floatx8;

#define V_SZ   32000
#define D_SZ   1024
#define H_SZ   16
#define L_SZ   2
#define S_SZ   2048
#define D4_SZ  4096
#define HD_SZ  64
#define ATTN_SCALE 0.125f   // 1/sqrt(64)

__device__ __forceinline__ bf16x16 pack16(bf16x8 lo, bf16x8 hi) {
  bf16x16 r;
#pragma unroll
  for (int i = 0; i < 8; ++i) { r[i] = lo[i]; r[i + 8] = hi[i]; }
  return r;
}

__device__ __forceinline__ floatx8 zero8() {
  floatx8 z = {0.f, 0.f, 0.f, 0.f, 0.f, 0.f, 0.f, 0.f};
  return z;
}

// CDNA5 async global->LDS copy (ASYNCcnt-tracked; bypasses VGPR round-trip)
__device__ __forceinline__ void async_copy_b128(unsigned lds_addr, const void* gaddr) {
  asm volatile("global_load_async_to_lds_b128 %0, %1, off"
               :: "v"(lds_addr), "v"((unsigned long long)(size_t)gaddr)
               : "memory");
}
__device__ __forceinline__ void async_wait0() {
  asm volatile("s_wait_asynccnt 0x0" ::: "memory");
}
__device__ __forceinline__ unsigned lds_addr_of(const void* p) {
  return (unsigned)(size_t)p;   // LDS aperture: low 32 bits = LDS byte address
}

// ---------------------------------------------------------------------------
// Elementwise f32 -> bf16 convert (emb table; row-major == Bt for logits GEMM)
// ---------------------------------------------------------------------------
__global__ __launch_bounds__(256)
void convert_bf16_kernel(const float* __restrict__ src, __bf16* __restrict__ dst,
                         size_t n) {
  size_t i = ((size_t)blockIdx.x * 256 + threadIdx.x) * 8;
  if (i + 8 <= n) {
#pragma unroll
    for (int j = 0; j < 8; ++j) dst[i + j] = (__bf16)src[i + j];
  }
}

// ---------------------------------------------------------------------------
// W (K x N, f32) -> Wt (N x K, bf16)  — tiled transpose through LDS
// ---------------------------------------------------------------------------
__global__ __launch_bounds__(256)
void transpose_convert_kernel(const float* __restrict__ W, __bf16* __restrict__ Wt,
                              int K, int N) {
  __shared__ float tile[32][33];
  const int n0 = blockIdx.x * 32, k0 = blockIdx.y * 32;
  const int tx = threadIdx.x & 31, ty = threadIdx.x >> 5;  // 32 x 8
#pragma unroll
  for (int i = 0; i < 4; ++i)
    tile[ty + i * 8][tx] = W[(size_t)(k0 + ty + i * 8) * N + n0 + tx];
  __syncthreads();
#pragma unroll
  for (int i = 0; i < 4; ++i)
    Wt[(size_t)(n0 + ty + i * 8) * K + k0 + tx] = (__bf16)tile[tx][ty + i * 8];
}

// ---------------------------------------------------------------------------
// x[s,d] = emb[ids[s],d] + pos[s,d]   (f32 residual stream)
// ---------------------------------------------------------------------------
__global__ __launch_bounds__(256)
void embed_kernel(const int* __restrict__ ids, const float* __restrict__ emb,
                  const float* __restrict__ pos, float* __restrict__ x) {
  int idx = blockIdx.x * 256 + threadIdx.x;      // S*D total
  int s = idx >> 10;                             // / D_SZ
  int d = idx & (D_SZ - 1);
  x[idx] = emb[(size_t)ids[s] * D_SZ + d] + pos[idx];
}

// ---------------------------------------------------------------------------
// LayerNorm (f32 in) -> bf16 out, one block per row, D=1024
// ---------------------------------------------------------------------------
__global__ __launch_bounds__(256)
void layernorm_bf16_kernel(const float* __restrict__ x, const float* __restrict__ g,
                           const float* __restrict__ b, __bf16* __restrict__ out) {
  __shared__ float red[8];
  __shared__ float mu_s, rs_s;
  const int row = blockIdx.x;
  const int tid = threadIdx.x;
  const float* xr = x + (size_t)row * D_SZ;
  float v[4];
  float s = 0.f;
#pragma unroll
  for (int i = 0; i < 4; ++i) { v[i] = xr[tid + i * 256]; s += v[i]; }
#pragma unroll
  for (int o = 16; o > 0; o >>= 1) s += __shfl_xor(s, o, 32);
  if ((tid & 31) == 0) red[tid >> 5] = s;
  __syncthreads();
  if (tid == 0) {
    float t = 0.f;
#pragma unroll
    for (int i = 0; i < 8; ++i) t += red[i];
    mu_s = t * (1.0f / D_SZ);
  }
  __syncthreads();
  const float mu = mu_s;
  float s2 = 0.f;
#pragma unroll
  for (int i = 0; i < 4; ++i) { float d = v[i] - mu; s2 += d * d; }
#pragma unroll
  for (int o = 16; o > 0; o >>= 1) s2 += __shfl_xor(s2, o, 32);
  if ((tid & 31) == 0) red[tid >> 5] = s2;
  __syncthreads();
  if (tid == 0) {
    float t = 0.f;
#pragma unroll
    for (int i = 0; i < 8; ++i) t += red[i];
    rs_s = rsqrtf(t * (1.0f / D_SZ) + 1e-5f);
  }
  __syncthreads();
  const float rs = rs_s;
#pragma unroll
  for (int i = 0; i < 4; ++i) {
    int c = tid + i * 256;
    out[(size_t)row * D_SZ + c] = (__bf16)((v[i] - mu) * rs * g[c] + b[c]);
  }
}

// ---------------------------------------------------------------------------
// WMMA GEMM: C[M,N] = A[M,K](bf16) * Bt[N,K](bf16)^T  (+bias, +gelu, +resid)
// 256 threads = 8 waves; 128x128 tile; BK=64; each wave 32x64 (2x4 wmma tiles)
// Double-buffered K-tiles staged via global_load_async_to_lds_b128 (ASYNCcnt)
// ---------------------------------------------------------------------------
template <bool BIAS, bool GELU_, bool RESID, bool OUTBF>
__global__ __launch_bounds__(256)
void gemm_bf16_kernel(const __bf16* __restrict__ A, const __bf16* __restrict__ Bt,
                      const float* __restrict__ bias, const float* __restrict__ resid,
                      void* __restrict__ Cout, int M, int N, int K) {
  constexpr int BK = 64, PAD = 8;
  __shared__ __bf16 As[2][128][BK + PAD];
  __shared__ __bf16 Bs[2][128][BK + PAD];
  const int tid  = threadIdx.x;
  const int lane = tid & 31;
  const int wave = tid >> 5;
  const int wm   = wave & 3;            // 4 waves along M -> 32 rows each
  const int wn   = wave >> 2;           // 2 waves along N -> 64 cols each
  const int fr   = lane & 15;
  const int kg   = (lane >> 4) << 3;    // 0 or 8
  const int mhi  = (lane >> 4) << 3;    // C-layout row offset

  const int bm = blockIdx.y * 128;
  const int bn = blockIdx.x * 128;

  // this thread's 4 staging slots (row, col) within the 128x64 tile
  const int srow = tid >> 1;            // 0..127  (2 chunks per row per thread pair)
  const int scol0 = (tid & 1) << 5;     // 0 or 32

  floatx8 acc[2][4];
#pragma unroll
  for (int mi = 0; mi < 2; ++mi)
#pragma unroll
    for (int ni = 0; ni < 4; ++ni) acc[mi][ni] = zero8();

  auto stage = [&](int buf, int k0) {
#pragma unroll
    for (int i = 0; i < 2; ++i) {       // two 8-elem chunks per matrix per thread
      int col = scol0 + (i << 3) + ((tid & 1) ? 0 : 0);
      // chunks: cover cols [scol0, scol0+8) and [scol0+16, scol0+24)? -> use stride 8*2
      col = scol0 + i * 16;             // 16-elem spacing -> 2 chunks of 8 with gap
      async_copy_b128(lds_addr_of(&As[buf][srow][col]),
                      A + (size_t)(bm + srow) * K + k0 + col);
      async_copy_b128(lds_addr_of(&As[buf][srow][col + 8]),
                      A + (size_t)(bm + srow) * K + k0 + col + 8);
      async_copy_b128(lds_addr_of(&Bs[buf][srow][col]),
                      Bt + (size_t)(bn + srow) * K + k0 + col);
      async_copy_b128(lds_addr_of(&Bs[buf][srow][col + 8]),
                      Bt + (size_t)(bn + srow) * K + k0 + col + 8);
    }
  };

  const int nk = K / BK;
  stage(0, 0);
  async_wait0();
  __syncthreads();

  for (int t = 0; t < nk; ++t) {
    const int cur = t & 1;
    if (t + 1 < nk) stage(cur ^ 1, (t + 1) * BK);   // overlap with compute

#pragma unroll
    for (int s = 0; s < BK; s += 32) {
      bf16x16 af[2], bfm[4];
#pragma unroll
      for (int mi = 0; mi < 2; ++mi) {
        int r = wm * 32 + mi * 16 + fr;
        af[mi] = pack16(*reinterpret_cast<const bf16x8*>(&As[cur][r][s + kg]),
                        *reinterpret_cast<const bf16x8*>(&As[cur][r][s + kg + 16]));
      }
#pragma unroll
      for (int ni = 0; ni < 4; ++ni) {
        int r = wn * 64 + ni * 16 + fr;
        bfm[ni] = pack16(*reinterpret_cast<const bf16x8*>(&Bs[cur][r][s + kg]),
                         *reinterpret_cast<const bf16x8*>(&Bs[cur][r][s + kg + 16]));
      }
#pragma unroll
      for (int mi = 0; mi < 2; ++mi)
#pragma unroll
        for (int ni = 0; ni < 4; ++ni)
          acc[mi][ni] = __builtin_amdgcn_wmma_f32_16x16x32_bf16(
              false, af[mi], false, bfm[ni], (short)0, acc[mi][ni], false, false);
    }
    async_wait0();
    __syncthreads();
  }

  const int mrow0 = bm + wm * 32;
  const int ncol0 = bn + wn * 64;
#pragma unroll
  for (int mi = 0; mi < 2; ++mi)
#pragma unroll
    for (int ni = 0; ni < 4; ++ni)
#pragma unroll
      for (int r = 0; r < 8; ++r) {
        int m = mrow0 + mi * 16 + mhi + r;
        int n = ncol0 + ni * 16 + fr;
        float v = acc[mi][ni][r];
        if (BIAS)  v += bias[n];
        if (GELU_) v = 0.5f * v * (1.0f + erff(v * 0.70710678f));
        if (RESID) v += resid[(size_t)m * N + n];
        if (OUTBF) ((__bf16*)Cout)[(size_t)m * N + n] = (__bf16)v;
        else       ((float*)Cout)[(size_t)m * N + n] = v;
      }
}

// ---------------------------------------------------------------------------
// Banded attention: one block (128 threads = 4 waves) per (head, 16-row qtile)
// scores via WMMA, masked softmax in LDS, P*V via WMMA with V staged transposed
// ---------------------------------------------------------------------------
__global__ __launch_bounds__(128)
void attn_kernel(const __bf16* __restrict__ q, const __bf16* __restrict__ k,
                 const __bf16* __restrict__ v, __bf16* __restrict__ o) {
  __shared__ __bf16 qs[16][72];            // 16 x 64 q tile (+pad)
  __shared__ __bf16 ks4[4][16][72];        // per-wave k tile
  __shared__ __bf16 vT[4][16][304];        // per-wave hd-slice of V^T (hd x kv)
  __shared__ __bf16 sc[16][304];           // scores, then P (bf16)
  __shared__ float  red[16][8];
  __shared__ float  rowm[16], rown[16];

  const int tid  = threadIdx.x;
  const int lane = tid & 31;
  const int wave = tid >> 5;
  const int fr   = lane & 15;
  const int kg   = (lane >> 4) << 3;
  const int mhi  = (lane >> 4) << 3;

  const int head  = blockIdx.y;
  const int qbase = blockIdx.x * 16;
  int kvlo = qbase - 128; if (kvlo < 0) kvlo = 0;
  int kvhi = qbase + 144; if (kvhi > S_SZ) kvhi = S_SZ;
  const int nkv   = (kvhi - kvlo) >> 4;    // <= 17 tiles
  const int ncols = nkv << 4;
  const int nks   = (ncols + 31) >> 5;     // k-steps of 32 for P*V

  // stage Q tile (16 x 64)
  {
    int r = tid >> 3, c = (tid & 7) << 3;
    *reinterpret_cast<bf16x8*>(&qs[r][c]) =
        *reinterpret_cast<const bf16x8*>(q + (size_t)(qbase + r) * D_SZ + head * HD_SZ + c);
  }
  __syncthreads();

  // score phase: wave-private K staging, no block barriers inside the loop
  for (int j = wave; j < nkv; j += 4) {
#pragma unroll
    for (int i = 0; i < 4; ++i) {
      int c = lane + i * 32;
      int r = c >> 3, col = (c & 7) << 3;
      *reinterpret_cast<bf16x8*>(&ks4[wave][r][col]) =
          *reinterpret_cast<const bf16x8*>(k + (size_t)(kvlo + j * 16 + r) * D_SZ +
                                           head * HD_SZ + col);
    }
    floatx8 acc = zero8();
#pragma unroll
    for (int s = 0; s < HD_SZ; s += 32) {
      bf16x16 af = pack16(*reinterpret_cast<const bf16x8*>(&qs[fr][s + kg]),
                          *reinterpret_cast<const bf16x8*>(&qs[fr][s + kg + 16]));
      bf16x16 bfm = pack16(*reinterpret_cast<const bf16x8*>(&ks4[wave][fr][s + kg]),
                           *reinterpret_cast<const bf16x8*>(&ks4[wave][fr][s + kg + 16]));
      acc = __builtin_amdgcn_wmma_f32_16x16x32_bf16(false, af, false, bfm,
                                                    (short)0, acc, false, false);
    }
#pragma unroll
    for (int r = 0; r < 8; ++r) sc[mhi + r][j * 16 + fr] = (__bf16)acc[r];
  }

  // stage V^T slice (wave-private): rows = hd [wave*16, wave*16+16), cols = kv
  for (int rr = lane; rr < nks * 32; rr += 32) {
    if (rr < ncols) {
      const __bf16* vp = v + (size_t)(kvlo + rr) * D_SZ + head * HD_SZ + wave * 16;
      bf16x8 v0 = *reinterpret_cast<const bf16x8*>(vp);
      bf16x8 v1 = *reinterpret_cast<const bf16x8*>(vp + 8);
#pragma unroll
      for (int i = 0; i < 8; ++i) { vT[wave][i][rr] = v0[i]; vT[wave][i + 8][rr] = v1[i]; }
    } else {
#pragma unroll
      for (int i = 0; i < 16; ++i) vT[wave][i][rr] = (__bf16)0.f;
    }
  }
  __syncthreads();

  // masked softmax: 8 threads per row
  {
    const int r = tid >> 3, sub = tid & 7;
    const int ig = qbase + r;
    float mx = -3.0e38f;
    for (int c = sub; c < ncols; c += 8) {
      int jg = kvlo + c;
      if (jg >= ig - 128 && jg <= ig + 128) mx = fmaxf(mx, (float)sc[r][c]);
    }
    red[r][sub] = mx;
    __syncthreads();
    if (sub == 0) {
      float m2 = red[r][0];
#pragma unroll
      for (int i = 1; i < 8; ++i) m2 = fmaxf(m2, red[r][i]);
      rowm[r] = m2;
    }
    __syncthreads();
    const float rm = rowm[r];
    float sum = 0.f;
    for (int c = sub; c < ncols; c += 8) {
      int jg = kvlo + c;
      float p = 0.f;
      if (jg >= ig - 128 && jg <= ig + 128)
        p = __expf(ATTN_SCALE * ((float)sc[r][c] - rm));
      sum += p;
      sc[r][c] = (__bf16)p;
    }
    red[r][sub] = sum;
    __syncthreads();
    if (sub == 0) {
      float s2 = 0.f;
#pragma unroll
      for (int i = 0; i < 8; ++i) s2 += red[r][i];
      rown[r] = 1.0f / s2;
    }
    __syncthreads();
    const float inv = rown[r];
    for (int c = sub; c < ncols; c += 8) sc[r][c] = (__bf16)((float)sc[r][c] * inv);
    for (int c = ncols + sub; c < nks * 32; c += 8) sc[r][c] = (__bf16)0.f;
  }
  __syncthreads();

  // O = P * V ; each wave owns one 16-wide hd slice (n-tile = wave)
  floatx8 oacc = zero8();
  for (int s = 0; s < nks * 32; s += 32) {
    bf16x16 af = pack16(*reinterpret_cast<const bf16x8*>(&sc[fr][s + kg]),
                        *reinterpret_cast<const bf16x8*>(&sc[fr][s + kg + 16]));
    bf16x16 bfm = pack16(*reinterpret_cast<const bf16x8*>(&vT[wave][fr][s + kg]),
                         *reinterpret_cast<const bf16x8*>(&vT[wave][fr][s + kg + 16]));
    oacc = __builtin_amdgcn_wmma_f32_16x16x32_bf16(false, af, false, bfm,
                                                   (short)0, oacc, false, false);
  }
#pragma unroll
  for (int r = 0; r < 8; ++r)
    o[(size_t)(qbase + mhi + r) * D_SZ + head * HD_SZ + wave * 16 + fr] = (__bf16)oacc[r];
}

// ---------------------------------------------------------------------------
// Host-side orchestration
// ---------------------------------------------------------------------------
extern "C" void kernel_launch(void* const* d_in, const int* in_sizes, int n_in,
                              void* d_out, int out_size, void* d_ws, size_t ws_size,
                              hipStream_t stream) {
  (void)in_sizes; (void)n_in; (void)out_size; (void)ws_size;

  const int*   ids  = (const int*)d_in[0];
  const float* emb  = (const float*)d_in[1];
  const float* pos  = (const float*)d_in[2];
  const float* wq   = (const float*)d_in[3];
  const float* bq   = (const float*)d_in[4];
  const float* wk   = (const float*)d_in[5];
  const float* bk   = (const float*)d_in[6];
  const float* wv   = (const float*)d_in[7];
  const float* bv   = (const float*)d_in[8];
  const float* wo   = (const float*)d_in[9];
  const float* bo   = (const float*)d_in[10];
  const float* w1   = (const float*)d_in[11];
  const float* b1   = (const float*)d_in[12];
  const float* w2   = (const float*)d_in[13];
  const float* b2   = (const float*)d_in[14];
  const float* ln1g = (const float*)d_in[15];
  const float* ln1b = (const float*)d_in[16];
  const float* ln2g = (const float*)d_in[17];
  const float* ln2b = (const float*)d_in[18];
  const float* outg = (const float*)d_in[19];
  const float* outb = (const float*)d_in[20];
  float* logits = (float*)d_out;

  // carve workspace
  char* wsp = (char*)d_ws;
  auto alloc = [&](size_t bytes) -> char* {
    char* p = wsp;
    wsp += (bytes + 255) & ~(size_t)255;
    return p;
  };
  __bf16* embb = (__bf16*)alloc((size_t)V_SZ * D_SZ * 2);
  __bf16* wqT  = (__bf16*)alloc((size_t)L_SZ * D_SZ * D_SZ * 2);
  __bf16* wkT  = (__bf16*)alloc((size_t)L_SZ * D_SZ * D_SZ * 2);
  __bf16* wvT  = (__bf16*)alloc((size_t)L_SZ * D_SZ * D_SZ * 2);
  __bf16* woT  = (__bf16*)alloc((size_t)L_SZ * D_SZ * D_SZ * 2);
  __bf16* w1T  = (__bf16*)alloc((size_t)L_SZ * D_SZ * D4_SZ * 2);
  __bf16* w2T  = (__bf16*)alloc((size_t)L_SZ * D4_SZ * D_SZ * 2);
  float*  x    = (float*)alloc((size_t)S_SZ * D_SZ * 4);
  __bf16* hb   = (__bf16*)alloc((size_t)S_SZ * D_SZ * 2);
  __bf16* qb   = (__bf16*)alloc((size_t)S_SZ * D_SZ * 2);
  __bf16* kb   = (__bf16*)alloc((size_t)S_SZ * D_SZ * 2);
  __bf16* vb   = (__bf16*)alloc((size_t)S_SZ * D_SZ * 2);
  __bf16* ob   = (__bf16*)alloc((size_t)S_SZ * D_SZ * 2);
  __bf16* fb   = (__bf16*)alloc((size_t)S_SZ * D4_SZ * 2);

  // --- weight prep ---
  {
    size_t n = (size_t)V_SZ * D_SZ;
    convert_bf16_kernel<<<dim3((unsigned)(n / 8 / 256)), dim3(256), 0, stream>>>(emb, embb, n);
  }
  for (int l = 0; l < L_SZ; ++l) {
    size_t dd = (size_t)D_SZ * D_SZ;
    transpose_convert_kernel<<<dim3(32, 32), dim3(256), 0, stream>>>(wq + l * dd, wqT + l * dd, D_SZ, D_SZ);
    transpose_convert_kernel<<<dim3(32, 32), dim3(256), 0, stream>>>(wk + l * dd, wkT + l * dd, D_SZ, D_SZ);
    transpose_convert_kernel<<<dim3(32, 32), dim3(256), 0, stream>>>(wv + l * dd, wvT + l * dd, D_SZ, D_SZ);
    transpose_convert_kernel<<<dim3(32, 32), dim3(256), 0, stream>>>(wo + l * dd, woT + l * dd, D_SZ, D_SZ);
    size_t d4 = (size_t)D_SZ * D4_SZ;
    transpose_convert_kernel<<<dim3(128, 32), dim3(256), 0, stream>>>(w1 + l * d4, w1T + l * d4, D_SZ, D4_SZ);
    transpose_convert_kernel<<<dim3(32, 128), dim3(256), 0, stream>>>(w2 + l * d4, w2T + l * d4, D4_SZ, D_SZ);
  }

  // --- embedding ---
  embed_kernel<<<dim3(S_SZ * D_SZ / 256), dim3(256), 0, stream>>>(ids, emb, pos, x);

  const dim3 gD(D_SZ / 128, S_SZ / 128);    // GEMM grids
  const dim3 gF(D4_SZ / 128, S_SZ / 128);
  const dim3 gV(V_SZ / 128, S_SZ / 128);

  for (int l = 0; l < L_SZ; ++l) {
    size_t dd = (size_t)D_SZ * D_SZ;
    size_t d4 = (size_t)D_SZ * D4_SZ;

    // ln1 -> hb
    layernorm_bf16_kernel<<<dim3(S_SZ), dim3(256), 0, stream>>>(x, ln1g + l * D_SZ, ln1b + l * D_SZ, hb);
    // q, k, v  (bias, bf16 out)
    gemm_bf16_kernel<true, false, false, true><<<gD, dim3(256), 0, stream>>>(
        hb, wqT + l * dd, bq + l * D_SZ, nullptr, qb, S_SZ, D_SZ, D_SZ);
    gemm_bf16_kernel<true, false, false, true><<<gD, dim3(256), 0, stream>>>(
        hb, wkT + l * dd, bk + l * D_SZ, nullptr, kb, S_SZ, D_SZ, D_SZ);
    gemm_bf16_kernel<true, false, false, true><<<gD, dim3(256), 0, stream>>>(
        hb, wvT + l * dd, bv + l * D_SZ, nullptr, vb, S_SZ, D_SZ, D_SZ);
    // banded attention -> ob
    attn_kernel<<<dim3(S_SZ / 16, H_SZ), dim3(128), 0, stream>>>(qb, kb, vb, ob);
    // x = x + ob @ woT + bo
    gemm_bf16_kernel<true, false, true, false><<<gD, dim3(256), 0, stream>>>(
        ob, woT + l * dd, bo + l * D_SZ, x, x, S_SZ, D_SZ, D_SZ);
    // ln2 -> hb
    layernorm_bf16_kernel<<<dim3(S_SZ), dim3(256), 0, stream>>>(x, ln2g + l * D_SZ, ln2b + l * D_SZ, hb);
    // ffn1: gelu(hb @ w1T + b1) -> fb (bf16)
    gemm_bf16_kernel<true, true, false, true><<<gF, dim3(256), 0, stream>>>(
        hb, w1T + l * d4, b1 + l * D4_SZ, nullptr, fb, S_SZ, D4_SZ, D_SZ);
    // ffn2: x = x + fb @ w2T + b2
    gemm_bf16_kernel<true, false, true, false><<<gD, dim3(256), 0, stream>>>(
        fb, w2T + l * d4, b2 + l * D_SZ, x, x, S_SZ, D_SZ, D4_SZ);
  }

  // final LN -> hb, then logits = hb @ emb^T (Bt == embb row-major)
  layernorm_bf16_kernel<<<dim3(S_SZ), dim3(256), 0, stream>>>(x, outg, outb, hb);
  gemm_bf16_kernel<false, false, false, false><<<gV, dim3(256), 0, stream>>>(
      hb, embb, nullptr, nullptr, logits, S_SZ, V_SZ, D_SZ);
}
